// HeterogeneousEdgeGraphSAGELSTMv1_44444321579085
// MI455X (gfx1250) — matrix operand order repaired and checked by reading, not compiled, not run on hardware
//
#include <hip/hip_runtime.h>

#define MAXDEG 12

typedef __bf16 bf16_t;
typedef bf16_t v16bf __attribute__((ext_vector_type(16)));
typedef float  v8f   __attribute__((ext_vector_type(8)));
typedef unsigned short u16;
typedef unsigned long long u64;

#if defined(__has_builtin)
#if __has_builtin(__builtin_amdgcn_global_load_async_to_lds_b128)
#define HAVE_ASYNC_LDS 1
#endif
#if __has_builtin(__builtin_amdgcn_s_wait_asynccnt)
#define HAVE_WAIT_ASYNC 1
#endif
#endif

union FragU { v16bf v; uint4 q[2]; };

__device__ __forceinline__ u16 f2bf(float f) {
    unsigned u = __float_as_uint(f);
    unsigned r = u + 0x7fffu + ((u >> 16) & 1u);
    return (u16)(r >> 16);
}
__device__ __forceinline__ float sigmoidf_(float x) { return 1.0f / (1.0f + __expf(-x)); }

__device__ __forceinline__ v8f wmma_bf16(v16bf a, v16bf b, v8f c) {
    return __builtin_amdgcn_wmma_f32_16x16x32_bf16(false, a, false, b, (short)0, c, false, false);
}

// A-fragment (16x32 bf16). lane row m = lane&15, kh = lane>>4.
// VGPR dwords 0-3 hold K pairs {4*kh+v}, dwords 4-7 hold pairs {8+4*kh+v}.
__device__ __forceinline__ v16bf load_a_frag(const void* rowBase, int kh, int kfrag) {
    FragU f;
    const char* p = (const char*)rowBase + kfrag * 64 + kh * 16;
    f.q[0] = *(const uint4*)(p);
    f.q[1] = *(const uint4*)(p + 32);
    return f.v;
}
// B-fragment (32x16 bf16). lane col n = lane&15, kh = lane>>4.
// elements 0-7 hold K=8*kh..8*kh+7, elements 8-15 hold K=16+8*kh..16+8*kh+7 (same as A).
// rowBase points at W[n_global][0] (row of 64 bf16, i.e. B column n).
__device__ __forceinline__ v16bf load_b_frag(const void* rowBase, int kh, int kfrag) {
    FragU f;
    const char* p = (const char*)rowBase + kfrag * 64 + kh * 16;
    f.q[0] = *(const uint4*)(p);
    f.q[1] = *(const uint4*)(p + 32);
    return f.v;
}

// ---------------- encoder: out = relu(x@W1.T+b1)@W2.T+b2 (+bf16 mirror) -------------
__global__ void enc_kernel(const float* __restrict__ x, int din,
                           const float* __restrict__ W1, const float* __restrict__ b1,
                           const float* __restrict__ W2, const float* __restrict__ b2,
                           float* __restrict__ out, u16* __restrict__ out_bf) {
    __shared__ float row[32];
    __shared__ float hid[64];
    int nidx = blockIdx.x, tid = threadIdx.x;
    if (tid < din) row[tid] = x[nidx * din + tid];
    __syncthreads();
    float a = b1[tid];
    for (int k = 0; k < din; ++k) a += row[k] * W1[tid * din + k];
    hid[tid] = fmaxf(a, 0.f);
    __syncthreads();
    float o = b2[tid];
    #pragma unroll 8
    for (int k = 0; k < 64; ++k) o += hid[k] * W2[tid * 64 + k];
    out[nidx * 64 + tid]    = o;
    out_bf[nidx * 64 + tid] = f2bf(o);
}

__global__ void zero_tail_kernel(u16* a, u16* b, int n) {
    int i = threadIdx.x;
    if (i < 64) { a[n * 64 + i] = 0; b[n * 64 + i] = 0; }
}

// ---------------- weight conversion f32 -> bf16, combined biases ----------------
__global__ void cvt_weights_kernel(const float* __restrict__ wih, const float* __restrict__ whh,
                                   const float* __restrict__ wl,  const float* __restrict__ wr,
                                   const float* __restrict__ bih, const float* __restrict__ bhh,
                                   u16* wih_bf, u16* whh_bf, u16* wl_bf, u16* wr_bf, float* bsum) {
    int i = blockIdx.x * blockDim.x + threadIdx.x;
    const int A = 12 * 256 * 64;
    const int B = 12 * 64 * 64;
    if (i < A)        { wih_bf[i] = f2bf(wih[i]); return; }
    i -= A;
    if (i < A)        { whh_bf[i] = f2bf(whh[i]); return; }
    i -= A;
    if (i < B)        { wl_bf[i] = f2bf(wl[i]); return; }
    i -= B;
    if (i < B)        { wr_bf[i] = f2bf(wr[i]); return; }
    i -= B;
    if (i < 12 * 256) { bsum[i] = bih[i] + bhh[i]; }
}

// ---------------- graph prep: CSR by dst, sorted top-12 neighbors ----------------
__global__ void init_deg_kernel(int* deg, int n) {
    int i = blockIdx.x * blockDim.x + threadIdx.x;
    if (i < n) deg[i] = 1;                     // self loop pre-counted
}
__global__ void count_deg_kernel(const int* __restrict__ dst, int* deg, int E) {
    int e = blockIdx.x * blockDim.x + threadIdx.x;
    if (e < E) atomicAdd(&deg[dst[e]], 1);
}
__global__ void scan_kernel(const int* __restrict__ deg, int* rowstart, int* cursor, int n) {
    __shared__ int buf[1024];
    __shared__ int carry;
    if (threadIdx.x == 0) carry = 0;
    __syncthreads();
    for (int base = 0; base < n; base += 1024) {
        int i = base + threadIdx.x;
        int v = (i < n) ? deg[i] : 0;
        buf[threadIdx.x] = v;
        __syncthreads();
        for (int off = 1; off < 1024; off <<= 1) {
            int t = (threadIdx.x >= off) ? buf[threadIdx.x - off] : 0;
            __syncthreads();
            buf[threadIdx.x] += t;
            __syncthreads();
        }
        int incl = buf[threadIdx.x];
        int excl = incl - v + carry;
        if (i < n) { rowstart[i] = excl; cursor[i] = excl; }
        __syncthreads();
        if (threadIdx.x == 1023) carry += incl;
        __syncthreads();
    }
}
__global__ void fill_edges_kernel(const int* __restrict__ src, const int* __restrict__ dst,
                                  int* cursor, u64* keys, int E, u64 KEYM) {
    int e = blockIdx.x * blockDim.x + threadIdx.x;
    if (e < E) {
        int p = atomicAdd(&cursor[dst[e]], 1);
        keys[p] = (u64)src[e] * KEYM + (u64)e;
    }
}
__global__ void fill_self_kernel(int* cursor, u64* keys, int n, int E, u64 KEYM) {
    int i = blockIdx.x * blockDim.x + threadIdx.x;
    if (i < n) {
        int p = atomicAdd(&cursor[i], 1);
        keys[p] = (u64)i * KEYM + (u64)(E + i);
    }
}
__global__ void build_nbr_kernel(const u64* __restrict__ keys, const int* __restrict__ rowstart,
                                 const int* __restrict__ deg, int* __restrict__ nbr,
                                 int n, u64 KEYM) {
    int d = blockIdx.x * blockDim.x + threadIdx.x;
    if (d >= n) return;
    int st = rowstart[d], dg = deg[d];
    u64 best[MAXDEG];
    int cnt = 0;
    for (int p = 0; p < dg; ++p) {
        u64 k = keys[st + p];
        if (cnt < MAXDEG) {
            int j = cnt++;
            while (j > 0 && best[j - 1] > k) { best[j] = best[j - 1]; --j; }
            best[j] = k;
        } else if (k < best[MAXDEG - 1]) {
            int j = MAXDEG - 1;
            while (j > 0 && best[j - 1] > k) { best[j] = best[j - 1]; --j; }
            best[j] = k;
        }
    }
    for (int t = 0; t < MAXDEG; ++t)
        nbr[d * MAXDEG + t] = (t < cnt) ? (int)(best[t] / KEYM) : n;   // n = zero row
}

// ---------------- the WMMA LSTM-SAGE kernel ----------------
// 2 waves / block, one 16-node tile per wave.
// LDS: [0,32768) Wih bf16 [256][64]; [32768,65536) Whh; htile per wave at 65536+wave*2048.
__global__ __launch_bounds__(64)
void lstm_sage_kernel(const u16* __restrict__ xsrc_bf, const u16* __restrict__ xdst_bf,
                      const int* __restrict__ nbr,
                      const u16* __restrict__ wih_bf, const u16* __restrict__ whh_bf,
                      const float* __restrict__ bsum,
                      const u16* __restrict__ wl_bf, const u16* __restrict__ wr_bf,
                      const float* __restrict__ bl,
                      float* __restrict__ out, int nnodes, int ntiles) {
    extern __shared__ char smem[];
#ifdef HAVE_ASYNC_LDS
    {   // async DMA of gate weights into LDS (GLOBAL_LOAD_ASYNC_TO_LDS_B128)
        typedef int v4i __attribute__((vector_size(16)));
        typedef __attribute__((address_space(1))) v4i gv4i;
        typedef __attribute__((address_space(3))) v4i lv4i;
        gv4i* g0 = (gv4i*)wih_bf;
        gv4i* g1 = (gv4i*)whh_bf;
        lv4i* l0 = (lv4i*)smem;
        lv4i* l1 = (lv4i*)(smem + 32768);
        for (int i = threadIdx.x; i < 2048; i += 64) {
            __builtin_amdgcn_global_load_async_to_lds_b128(g0 + i, l0 + i, 0, 0);
            __builtin_amdgcn_global_load_async_to_lds_b128(g1 + i, l1 + i, 0, 0);
        }
#ifdef HAVE_WAIT_ASYNC
        __builtin_amdgcn_s_wait_asynccnt(0);
#else
        asm volatile("s_wait_asynccnt 0" ::: "memory");
#endif
    }
#else
    {   // fallback: stage gate weights via VGPRs
        const uint4* s0 = (const uint4*)wih_bf;
        const uint4* s1 = (const uint4*)whh_bf;
        uint4* l0 = (uint4*)smem;
        uint4* l1 = (uint4*)(smem + 32768);
        for (int i = threadIdx.x; i < 2048; i += 64) l0[i] = s0[i];
        for (int i = threadIdx.x; i < 2048; i += 64) l1[i] = s1[i];
    }
#endif
    __syncthreads();

    int wave = threadIdx.x >> 5;
    int lane = threadIdx.x & 31;
    int tile = blockIdx.x * 2 + wave;
    if (tile >= ntiles) return;

    int nidx = lane & 15;      // A-row / B-col / C-col index
    int kh   = lane >> 4;
    int node_m  = tile * 16 + nidx;
    int node_ok = node_m < nnodes;
    char* hbase = smem + 65536 + wave * 2048;

    {   // zero h tile (16x64 bf16)
        uint4 zz; zz.x = zz.y = zz.z = zz.w = 0u;
        #pragma unroll
        for (int i = 0; i < 4; ++i) *(uint4*)(hbase + lane * 64 + i * 16) = zz;
    }
    asm volatile("s_wait_dscnt 0" ::: "memory");

    float barr[16];
    #pragma unroll
    for (int j = 0; j < 16; ++j) barr[j] = bsum[j * 16 + nidx];

    v8f c[4];
    #pragma unroll
    for (int q = 0; q < 4; ++q)
        #pragma unroll
        for (int r = 0; r < 8; ++r) c[q][r] = 0.f;

    // two-deep software pipeline on the neighbor gather:
    // x fragments for step t are loaded during step t-1; nbr index one further ahead.
    const int* nrow = nbr + (size_t)node_m * MAXDEG;
    int nb_cur = node_ok ? nrow[0] : nnodes;
    int nb_nxt = node_ok ? nrow[1] : nnodes;
    v16bf ax0 = load_a_frag(xsrc_bf + (size_t)nb_cur * 64, kh, 0);
    v16bf ax1 = load_a_frag(xsrc_bf + (size_t)nb_cur * 64, kh, 1);

    #pragma unroll 1
    for (int t = 0; t < MAXDEG; ++t) {
        // prefetch x for t+1 (index already resident) and nbr index for t+2
        v16bf axn0 = load_a_frag(xsrc_bf + (size_t)nb_nxt * 64, kh, 0);
        v16bf axn1 = load_a_frag(xsrc_bf + (size_t)nb_nxt * 64, kh, 1);
        nb_nxt = (node_ok && (t + 2 < MAXDEG)) ? nrow[t + 2] : nnodes;

        v16bf ah0 = load_a_frag(hbase + nidx * 128, kh, 0);
        v16bf ah1 = load_a_frag(hbase + nidx * 128, kh, 1);
        #pragma unroll
        for (int q = 0; q < 4; ++q) {           // hidden-unit chunk
            v8f acc[4];
            #pragma unroll
            for (int gi = 0; gi < 4; ++gi) {    // i, f, g, o
                int j = gi * 4 + q;             // gate n-tile 0..15
                v8f a;
                #pragma unroll
                for (int r = 0; r < 8; ++r) a[r] = barr[j];
                const char* bi = smem + (size_t)(j * 16 + nidx) * 128;
                a = wmma_bf16(ax0, load_b_frag(bi, kh, 0), a);
                a = wmma_bf16(ax1, load_b_frag(bi, kh, 1), a);
                const char* bh = smem + 32768 + (size_t)(j * 16 + nidx) * 128;
                a = wmma_bf16(ah0, load_b_frag(bh, kh, 0), a);
                a = wmma_bf16(ah1, load_b_frag(bh, kh, 1), a);
                acc[gi] = a;
            }
            #pragma unroll
            for (int r = 0; r < 8; ++r) {       // elementwise LSTM cell (C layout)
                float ig = sigmoidf_(acc[0][r]);
                float fg = sigmoidf_(acc[1][r]);
                float gg = tanhf(acc[2][r]);
                float og = sigmoidf_(acc[3][r]);
                float cn = fg * c[q][r] + ig * gg;
                c[q][r] = cn;
                float hh = og * tanhf(cn);
                *(u16*)(hbase + ((r + 8 * kh) * 64 + q * 16 + nidx) * 2) = f2bf(hh);
            }
        }
        ax0 = axn0;
        ax1 = axn1;
        asm volatile("s_wait_dscnt 0" ::: "memory");
    }

    // epilogue: out = h @ Wl.T + bl + x_dst @ Wr.T
    v16bf ah0 = load_a_frag(hbase + nidx * 128, kh, 0);
    v16bf ah1 = load_a_frag(hbase + nidx * 128, kh, 1);
    int dn = node_ok ? node_m : nnodes;
    v16bf ad0 = load_a_frag(xdst_bf + (size_t)dn * 64, kh, 0);
    v16bf ad1 = load_a_frag(xdst_bf + (size_t)dn * 64, kh, 1);
    #pragma unroll
    for (int q = 0; q < 4; ++q) {
        v8f a;
        float bb = bl[q * 16 + nidx];
        #pragma unroll
        for (int r = 0; r < 8; ++r) a[r] = bb;
        const u16* lrow = wl_bf + (size_t)(q * 16 + nidx) * 64;
        a = wmma_bf16(ah0, load_b_frag(lrow, kh, 0), a);
        a = wmma_bf16(ah1, load_b_frag(lrow, kh, 1), a);
        const u16* rrow = wr_bf + (size_t)(q * 16 + nidx) * 64;
        a = wmma_bf16(ad0, load_b_frag(rrow, kh, 0), a);
        a = wmma_bf16(ad1, load_b_frag(rrow, kh, 1), a);
        #pragma unroll
        for (int r = 0; r < 8; ++r) {
            int rowN = tile * 16 + r + 8 * kh;
            if (rowN < nnodes) out[(size_t)rowN * 64 + q * 16 + nidx] = a[r];
        }
    }
}

// ---------------- HeteroConv mean + residual + relu (+bf16 mirror) ----------------
__global__ void combine_kernel(const float* __restrict__ A, const float* __restrict__ B,
                               float* __restrict__ cur, u16* __restrict__ bf,
                               int total, int use_res) {
    int i = blockIdx.x * blockDim.x + threadIdx.x;
    if (i >= total) return;
    float v = 0.5f * (A[i] + B[i]) + (use_res ? cur[i] : 0.f);
    v = fmaxf(v, 0.f);
    cur[i] = v;
    bf[i]  = f2bf(v);
}

// ---------------- edge head ----------------
__global__ void edge_head_kernel(const float* __restrict__ s3, const float* __restrict__ t3,
                                 const int* __restrict__ esrc, const int* __restrict__ edst,
                                 const float* __restrict__ ea,
                                 const float* __restrict__ Wg1, const float* __restrict__ bg1,
                                 const float* __restrict__ Wg2, const float* __restrict__ bg2,
                                 const float* __restrict__ Wm1, const float* __restrict__ bm1,
                                 const float* __restrict__ Wm2, const float* __restrict__ bm2,
                                 const float* __restrict__ Wm3, const float* __restrict__ bm3,
                                 float* __restrict__ out, int E) {
    __shared__ float cat[128];
    __shared__ float red[64];
    __shared__ float er80[80];
    __shared__ float h1[64];
    __shared__ float h2[32];
    __shared__ float gsh;
    int e = blockIdx.x, tid = threadIdx.x;
    int s = esrc[e], d = edst[e];
    cat[tid]      = s3[(size_t)s * 64 + tid];
    cat[64 + tid] = t3[(size_t)d * 64 + tid];
    __syncthreads();
    float a = bg1[tid];
    #pragma unroll 8
    for (int k = 0; k < 128; ++k) a += cat[k] * Wg1[tid * 128 + k];
    red[tid] = fmaxf(a, 0.f) * Wg2[tid];
    __syncthreads();
    for (int off = 32; off > 0; off >>= 1) {
        if (tid < off) red[tid] += red[tid + off];
        __syncthreads();
    }
    if (tid == 0) gsh = sigmoidf_(red[0] + bg2[0]);
    __syncthreads();
    float g = gsh;
    er80[tid] = g * cat[tid] + (1.f - g) * cat[64 + tid];
    if (tid < 16) er80[64 + tid] = ea[(size_t)e * 16 + tid];
    __syncthreads();
    float m = bm1[tid];
    #pragma unroll 8
    for (int k = 0; k < 80; ++k) m += er80[k] * Wm1[tid * 80 + k];
    h1[tid] = fmaxf(m, 0.f);
    __syncthreads();
    if (tid < 32) {
        float v = bm2[tid];
        #pragma unroll 8
        for (int k = 0; k < 64; ++k) v += h1[k] * Wm2[tid * 64 + k];
        h2[tid] = fmaxf(v, 0.f);
    }
    __syncthreads();
    if (tid < 2) {
        float v = bm3[tid];
        #pragma unroll
        for (int k = 0; k < 32; ++k) v += h2[k] * Wm3[tid * 32 + k];
        out[(size_t)e * 2 + tid] = v;
    }
}

// ---------------- host orchestration ----------------
extern "C" void kernel_launch(void* const* d_in, const int* in_sizes, int n_in,
                              void* d_out, int out_size, void* d_ws, size_t ws_size,
                              hipStream_t stream) {
    (void)n_in; (void)out_size; (void)ws_size;
    const float* x_source  = (const float*)d_in[0];
    const float* x_target  = (const float*)d_in[1];
    const int*   edges     = (const int*)  d_in[2];
    const float* edge_attr = (const float*)d_in[3];
    const float* We_s1 = (const float*)d_in[4];  const float* be_s1 = (const float*)d_in[5];
    const float* We_s2 = (const float*)d_in[6];  const float* be_s2 = (const float*)d_in[7];
    const float* We_t1 = (const float*)d_in[8];  const float* be_t1 = (const float*)d_in[9];
    const float* We_t2 = (const float*)d_in[10]; const float* be_t2 = (const float*)d_in[11];
    const float* Wih = (const float*)d_in[12];   const float* Whh = (const float*)d_in[13];
    const float* bih = (const float*)d_in[14];   const float* bhh = (const float*)d_in[15];
    const float* Wl  = (const float*)d_in[16];   const float* blp = (const float*)d_in[17];
    const float* Wr  = (const float*)d_in[18];
    const float* Wg1 = (const float*)d_in[19];   const float* bg1 = (const float*)d_in[20];
    const float* Wg2 = (const float*)d_in[21];   const float* bg2 = (const float*)d_in[22];
    const float* Wm1 = (const float*)d_in[23];   const float* bm1 = (const float*)d_in[24];
    const float* Wm2 = (const float*)d_in[25];   const float* bm2 = (const float*)d_in[26];
    const float* Wm3 = (const float*)d_in[27];   const float* bm3 = (const float*)d_in[28];

    const int N = in_sizes[0] / 32;   // nodes per type
    const int E = in_sizes[2] / 8;    // edges per edge type
    const u64 KEYM = (u64)(E + N + 1);

    // bump allocator over workspace
    char* wsp = (char*)d_ws;
    size_t off = 0;
    auto alloc = [&](size_t bytes) -> void* {
        void* p = wsp + off;
        off += (bytes + 255) & ~(size_t)255;
        return p;
    };
    float* xs  = (float*)alloc((size_t)N * 64 * 4);
    float* xt  = (float*)alloc((size_t)N * 64 * 4);
    float* oAs = (float*)alloc((size_t)N * 64 * 4);
    float* oBs = (float*)alloc((size_t)N * 64 * 4);
    float* oAt = (float*)alloc((size_t)N * 64 * 4);
    float* oBt = (float*)alloc((size_t)N * 64 * 4);
    u16* xs_bf = (u16*)alloc((size_t)(N + 1) * 64 * 2);   // row N = zeros
    u16* xt_bf = (u16*)alloc((size_t)(N + 1) * 64 * 2);
    u16* wih_bf = (u16*)alloc((size_t)12 * 16384 * 2);
    u16* whh_bf = (u16*)alloc((size_t)12 * 16384 * 2);
    u16* wl_bf  = (u16*)alloc((size_t)12 * 4096 * 2);
    u16* wr_bf  = (u16*)alloc((size_t)12 * 4096 * 2);
    float* bsum = (float*)alloc((size_t)12 * 256 * 4);
    int* nbr      = (int*)alloc((size_t)4 * N * MAXDEG * 4);
    int* deg      = (int*)alloc((size_t)N * 4);
    int* rowstart = (int*)alloc((size_t)N * 4);
    int* cursor   = (int*)alloc((size_t)N * 4);
    u64* keys     = (u64*)alloc((size_t)(E + N) * 8);

    // 1) encoders (+ bf16 mirrors) and zero row
    enc_kernel<<<N, 64, 0, stream>>>(x_source, 32, We_s1, be_s1, We_s2, be_s2, xs, xs_bf);
    enc_kernel<<<N, 64, 0, stream>>>(x_target, 24, We_t1, be_t1, We_t2, be_t2, xt, xt_bf);
    zero_tail_kernel<<<1, 64, 0, stream>>>(xs_bf, xt_bf, N);

    // 2) bf16 weights + combined biases
    cvt_weights_kernel<<<(494592 + 255) / 256, 256, 0, stream>>>(
        Wih, Whh, Wl, Wr, bih, bhh, wih_bf, whh_bf, wl_bf, wr_bf, bsum);

    // 3) graph preprocessing per edge type (fixed across layers)
    for (int et = 0; et < 4; ++et) {
        const int* esrc = edges + (size_t)(et * 2 + 0) * E;
        const int* edst = edges + (size_t)(et * 2 + 1) * E;
        init_deg_kernel<<<(N + 255) / 256, 256, 0, stream>>>(deg, N);
        count_deg_kernel<<<(E + 255) / 256, 256, 0, stream>>>(edst, deg, E);
        scan_kernel<<<1, 1024, 0, stream>>>(deg, rowstart, cursor, N);
        fill_edges_kernel<<<(E + 255) / 256, 256, 0, stream>>>(esrc, edst, cursor, keys, E, KEYM);
        fill_self_kernel<<<(N + 255) / 256, 256, 0, stream>>>(cursor, keys, N, E, KEYM);
        build_nbr_kernel<<<(N + 255) / 256, 256, 0, stream>>>(
            keys, rowstart, deg, nbr + (size_t)et * N * MAXDEG, N, KEYM);
    }

    // 4) 3 layers x 4 SAGE-LSTM blocks (WMMA) + hetero-mean combine
    const int ntiles = (N + 15) / 16;
    dim3 lgrid((ntiles + 1) / 2);
    const size_t dynLds = 65536 + 2 * 2048;
    const u16* srcs[4]  = { xs_bf, xt_bf, xs_bf, xt_bf };
    const u16* dstsv[4] = { xs_bf, xt_bf, xt_bf, xs_bf };
    float* convs[4]     = { oAs,   oAt,   oBt,   oBs   };
    for (int L = 0; L < 3; ++L) {
        for (int et = 0; et < 4; ++et) {
            int i = L * 4 + et;
            lstm_sage_kernel<<<lgrid, 64, dynLds, stream>>>(
                srcs[et], dstsv[et], nbr + (size_t)et * N * MAXDEG,
                wih_bf + (size_t)i * 16384, whh_bf + (size_t)i * 16384,
                bsum + (size_t)i * 256,
                wl_bf + (size_t)i * 4096, wr_bf + (size_t)i * 4096,
                blp + (size_t)i * 64,
                convs[et], N, ntiles);
        }
        int use_res = (L > 0) ? 1 : 0;
        combine_kernel<<<(N * 64 + 255) / 256, 256, 0, stream>>>(oAs, oBs, xs, xs_bf, N * 64, use_res);
        combine_kernel<<<(N * 64 + 255) / 256, 256, 0, stream>>>(oAt, oBt, xt, xt_bf, N * 64, use_res);
    }

    // 5) edge head on original src->tgt edges
    edge_head_kernel<<<E, 64, 0, stream>>>(
        xs, xt, edges + (size_t)(2 * 2 + 0) * E, edges + (size_t)(2 * 2 + 1) * E, edge_attr,
        Wg1, bg1, Wg2, bg2, Wm1, bm1, Wm2, bm2, Wm3, bm3, (float*)d_out, E);
}
